// MSAColAttention_37417755083155
// MI455X (gfx1250) — compile-verified, matrix operand order checked by reading, not run on hardware
//
#include <hip/hip_runtime.h>

#define DM    256
#define NSEQ  256
#define LCOL  384
#define NHEAD 8
#define DHID  32
#define HD    (NHEAD * DHID)          // 256
#define MROWS (NSEQ * LCOL)           // 98304

typedef __attribute__((ext_vector_type(16))) _Float16     v16h;
typedef __attribute__((ext_vector_type(8)))  _Float16     h8;
typedef __attribute__((ext_vector_type(8)))  float        v8f;
typedef __attribute__((ext_vector_type(4)))  unsigned int v4u;
typedef __attribute__((ext_vector_type(8)))  int          v8i;
typedef __attribute__((ext_vector_type(4)))  int          v4i;

#if defined(__has_builtin)
#if __has_builtin(__builtin_amdgcn_tensor_load_to_lds) && \
    __has_builtin(__builtin_amdgcn_s_wait_tensorcnt)
#define USE_TDM 1
#endif
#endif
#ifndef USE_TDM
#define USE_TDM 0
#endif

static __device__ inline v8f wmma_f16(v16h a, v16h b, v8f c) {
  // (neg_a, A, neg_b, B, c_mod, C, reuse_a, reuse_b)
  return __builtin_amdgcn_wmma_f32_16x16x32_f16(false, a, false, b, (short)0, c,
                                                false, false);
}

// 16-bit A/B fragment when the lane's row (A: M, B: N) is contiguous in memory.
// Halves cover K = kb..kb+7 and kb+16..kb+23, kb = (lane>>4)*8  (ISA 7.12.2).
static __device__ inline v16h frag_rowmajor(const _Float16* rowptr, int lane) {
  const int kb = (lane >> 4) * 8;
  h8 lo = *(const h8*)(rowptr + kb);
  h8 hi = *(const h8*)(rowptr + kb + 16);
  v16h f;
#pragma unroll
  for (int j = 0; j < 8; ++j) { f[j] = lo[j]; f[j + 8] = hi[j]; }
  return f;
}

// Same fragment pattern gathered from an LDS tile with row `stride` (halves).
static __device__ inline v16h frag_lds(const _Float16* base, int row_base,
                                       int stride, int lane) {
  const int r  = row_base + (lane & 15);
  const int kb = (lane >> 4) * 8;
  const _Float16* p = base + r * stride;
  v16h f;
#pragma unroll
  for (int j = 0; j < 8; ++j) { f[j] = p[kb + j]; f[j + 8] = p[kb + 16 + j]; }
  return f;
}

// B fragment when data is stored K-major ([k][n] rows): lane owns column n.
static __device__ inline v16h frag_colmajor(const _Float16* base, int colbase,
                                            int stride, int lane) {
  const int c  = colbase + (lane & 15);
  const int kb = (lane >> 4) * 8;
  v16h f;
#pragma unroll
  for (int j = 0; j < 8; ++j) {
    f[j]     = base[(kb + j) * stride + c];
    f[j + 8] = base[(kb + 16 + j) * stride + c];
  }
  return f;
}

#if USE_TDM
// TDM: DMA a 2-D f16 tile (rowlen x nrows, row stride = rowlen elements,
// fully covering the tensor) from global memory into LDS. ISA 08 §8.3/8.4.
static __device__ inline void tdm_load_2d_f16(unsigned int lds_off,
                                              const _Float16* gptr,
                                              unsigned int rowlen,
                                              unsigned int nrows) {
  const unsigned long long ga = (unsigned long long)(uintptr_t)gptr;
  v4u g0;
  g0[0] = 1u;                                   // count=1, user mode
  g0[1] = lds_off;                              // lds_addr
  g0[2] = (unsigned int)(ga & 0xffffffffu);     // global_addr[31:0]
  g0[3] = ((unsigned int)(ga >> 32) & 0x01ffffffu) | 0x80000000u; // addr[56:32]|type=2
  v8i g1;
  g1[0] = 0x00010000;                           // workgroup_mask=0, data_size=2B
  g1[1] = (int)(rowlen << 16);                  // tensor_dim0[15:0]
  g1[2] = (int)(nrows << 16);                   // tensor_dim1[15:0] (dim0 hi = 0)
  g1[3] = (int)(rowlen << 16);                  // tile_dim0 (dim1 hi = 0)
  g1[4] = (int)nrows;                           // tile_dim1 (tile_dim2 = 0)
  g1[5] = (int)rowlen;                          // tensor_dim0_stride[31:0]
  g1[6] = 0;                                    // stride hi / dim1_stride lo
  g1[7] = 0;
  v4i gz = {0, 0, 0, 0};                        // groups 2/3: 2-D tensor
#if defined(__clang_major__) && (__clang_major__ >= 23)
  v8i gz8 = {0, 0, 0, 0, 0, 0, 0, 0};
  __builtin_amdgcn_tensor_load_to_lds(g0, g1, gz, gz, gz8, 0);
#else
  __builtin_amdgcn_tensor_load_to_lds(g0, g1, gz, gz, 0);
#endif
}
#endif

// ---------------------------------------------------------------- LayerNorm
__global__ __launch_bounds__(256) void msa_ln_kernel(
    const float* __restrict__ msa, const float* __restrict__ lnw,
    const float* __restrict__ lnb, _Float16* __restrict__ xo) {
  const int lane = threadIdx.x & 31;
  const int wid  = threadIdx.x >> 5;
  const long long row = (long long)blockIdx.x * 8 + wid;
  const float* src = msa + row * DM;
  float4 a = ((const float4*)src)[lane * 2];
  float4 b = ((const float4*)src)[lane * 2 + 1];
  float s  = a.x + a.y + a.z + a.w + b.x + b.y + b.z + b.w;
  float ss = a.x*a.x + a.y*a.y + a.z*a.z + a.w*a.w +
             b.x*b.x + b.y*b.y + b.z*b.z + b.w*b.w;
#pragma unroll
  for (int off = 16; off >= 1; off >>= 1) {
    s  += __shfl_xor(s,  off, 32);
    ss += __shfl_xor(ss, off, 32);
  }
  const float mu   = s * (1.0f / DM);
  const float rstd = rsqrtf(ss * (1.0f / DM) - mu * mu + 1e-5f);
  float4 w0 = ((const float4*)lnw)[lane * 2], w1 = ((const float4*)lnw)[lane * 2 + 1];
  float4 c0 = ((const float4*)lnb)[lane * 2], c1 = ((const float4*)lnb)[lane * 2 + 1];
  h8 o;
  o[0] = (_Float16)((a.x - mu) * rstd * w0.x + c0.x);
  o[1] = (_Float16)((a.y - mu) * rstd * w0.y + c0.y);
  o[2] = (_Float16)((a.z - mu) * rstd * w0.z + c0.z);
  o[3] = (_Float16)((a.w - mu) * rstd * w0.w + c0.w);
  o[4] = (_Float16)((b.x - mu) * rstd * w1.x + c1.x);
  o[5] = (_Float16)((b.y - mu) * rstd * w1.y + c1.y);
  o[6] = (_Float16)((b.z - mu) * rstd * w1.z + c1.z);
  o[7] = (_Float16)((b.w - mu) * rstd * w1.w + c1.w);
  *(h8*)(xo + row * DM + lane * 8) = o;
}

// --------------------------------------------------- fused Q/K/V/G projection
__global__ __launch_bounds__(256) void msa_proj_kernel(
    const _Float16* __restrict__ x,
    const float* __restrict__ wq, const float* __restrict__ wk,
    const float* __restrict__ wvm, const float* __restrict__ wg,
    const float* __restrict__ bg,
    _Float16* __restrict__ qo, _Float16* __restrict__ ko,
    _Float16* __restrict__ vo, _Float16* __restrict__ go) {
  __shared__ _Float16 xs[64][40];
  __shared__ _Float16 wt[64][40];
  const int tid   = threadIdx.x;
  const int lane  = tid & 31;
  const int wid   = tid >> 5;
  const int mbase = blockIdx.x * 64;
  const int mat   = blockIdx.y >> 2;            // 0:q 1:k 2:v 3:g
  const int colt  = (blockIdx.y & 3) * 64;
  const float* W = (mat == 0) ? wq : (mat == 1) ? wk : (mat == 2) ? wvm : wg;
  const int lrow = tid >> 2;                    // 0..63
  const int lseg = (tid & 3) * 8;               // 0,8,16,24
  const int frow = (wid >> 1) * 16;             // wave's 16-row chunk
  const int fcol = (wid & 1) * 32;              // wave's 32-col half
  v8f acc0 = {}, acc1 = {};
  for (int kb = 0; kb < DM; kb += 32) {
    __syncthreads();
    {   // stage X tile (f16), 128-bit global loads
      h8 d = *(const h8*)(x + (size_t)(mbase + lrow) * DM + kb + lseg);
#pragma unroll
      for (int j = 0; j < 8; ++j) xs[lrow][lseg + j] = d[j];
    }
    {   // stage W tile, f32 -> f16
      const float* wp = W + (size_t)(colt + lrow) * DM + kb + lseg;
      float4 d0 = ((const float4*)wp)[0];
      float4 d1 = ((const float4*)wp)[1];
      wt[lrow][lseg + 0] = (_Float16)d0.x; wt[lrow][lseg + 1] = (_Float16)d0.y;
      wt[lrow][lseg + 2] = (_Float16)d0.z; wt[lrow][lseg + 3] = (_Float16)d0.w;
      wt[lrow][lseg + 4] = (_Float16)d1.x; wt[lrow][lseg + 5] = (_Float16)d1.y;
      wt[lrow][lseg + 6] = (_Float16)d1.z; wt[lrow][lseg + 7] = (_Float16)d1.w;
    }
    __syncthreads();
    v16h a  = frag_lds(&xs[0][0], frow,      40, lane);
    v16h b0 = frag_lds(&wt[0][0], fcol,      40, lane);
    v16h b1 = frag_lds(&wt[0][0], fcol + 16, 40, lane);
    acc0 = wmma_f16(a, b0, acc0);
    acc1 = wmma_f16(a, b1, acc1);
  }
  const int lh = lane >> 4;
  const int lc = lane & 15;
#pragma unroll
  for (int r = 0; r < 8; ++r) {
    const int m     = mbase + frow + r + lh * 8;
    const int n_idx = m / LCOL;
    const int l_idx = m % LCOL;
#pragma unroll
    for (int s = 0; s < 2; ++s) {
      const int c = colt + fcol + s * 16 + lc;
      float val = (s == 0) ? acc0[r] : acc1[r];
      if (mat == 3) {
        const float g = 1.0f / (1.0f + __expf(-(val + bg[c])));
        go[(size_t)m * HD + c] = (_Float16)g;
      } else {
        if (mat == 0) val *= 0.17677669529663687f;  // 1/sqrt(32)
        _Float16* dst = (mat == 0) ? qo : (mat == 1) ? ko : vo;
        const int hh = c >> 5, dd = c & 31;
        dst[(((size_t)l_idx * NHEAD + hh) * NSEQ + n_idx) * DHID + dd] =
            (_Float16)val;
      }
    }
  }
}

// ----------------------------------------------- flash attention per (l, head)
__global__ __launch_bounds__(256) void msa_attn_kernel(
    const _Float16* __restrict__ q, const _Float16* __restrict__ k,
    const _Float16* __restrict__ v, const _Float16* __restrict__ gate,
    _Float16* __restrict__ att) {
  __shared__ _Float16 pbuf[8][32][40];      // per-wave P transpose staging
  __shared__ _Float16 vsbuf[NSEQ][DHID];    // whole V tensor for this (l,h), 16KB
  const int lane = threadIdx.x & 31;
  const int wid  = threadIdx.x >> 5;
  const int lh   = lane >> 4;
  const int lc   = lane & 15;
  const int li   = blockIdx.x >> 3;     // column index 0..383
  const int hh   = blockIdx.x & 7;      // head
  const size_t base = ((size_t)li * NHEAD + hh) * NSEQ * DHID;
  const _Float16* qb = q + base;
  const _Float16* kp = k + base;
  const _Float16* vp = v + base;
  const int row0 = wid * 32;            // this wave's 32 query rows

#if USE_TDM
  // One Tensor Data Mover DMA stages all of V (256x32 f16) into LDS while we
  // fetch Q fragments; wave 0 owns the descriptor, waits on TENSORcnt, then
  // the workgroup barrier publishes LDS to all waves.
  if (wid == 0) {
    tdm_load_2d_f16((unsigned int)(uintptr_t)&vsbuf[0][0], vp, DHID, NSEQ);
  }
#else
  // Fallback: cooperative staged copy of V into LDS (128-bit loads).
  for (int rr = wid * 4; rr < NSEQ; rr += 32) {
    const int r2 = rr + (lane >> 3);
    const int c2 = (lane & 7) * 4;
    *(h8*)&vsbuf[r2][c2] = *(const h8*)(vp + (size_t)r2 * DHID + c2);
  }
#endif

  const v16h qa0 = frag_rowmajor(qb + (size_t)(row0 + lc) * DHID,      lane);
  const v16h qa1 = frag_rowmajor(qb + (size_t)(row0 + 16 + lc) * DHID, lane);

#if USE_TDM
  if (wid == 0) __builtin_amdgcn_s_wait_tensorcnt(0);
#endif
  __syncthreads();

  v8f o00 = {}, o01 = {}, o10 = {}, o11 = {};
  float mrow[2][8], lrow[2][8];
#pragma unroll
  for (int c = 0; c < 2; ++c)
#pragma unroll
    for (int r = 0; r < 8; ++r) { mrow[c][r] = -1e30f; lrow[c][r] = 0.0f; }

  for (int kt = 0; kt < NSEQ; kt += 32) {
    if (kt + 32 < NSEQ) {
      __builtin_prefetch(kp + (size_t)(kt + 32) * DHID, 0, 1);
    }
    // S = Q @ K^T: B fragment rows are contiguous K rows (lane owns a key)
    const v16h kb0 = frag_rowmajor(kp + (size_t)(kt + lc) * DHID,      lane);
    const v16h kb1 = frag_rowmajor(kp + (size_t)(kt + 16 + lc) * DHID, lane);
    v8f zero = {};
    v8f s00 = wmma_f16(qa0, kb0, zero);
    v8f s01 = wmma_f16(qa0, kb1, zero);
    v8f s10 = wmma_f16(qa1, kb0, zero);
    v8f s11 = wmma_f16(qa1, kb1, zero);
    // online softmax (rows live in 16-lane halves; xor<=8 stays in-half)
#pragma unroll
    for (int c = 0; c < 2; ++c) {
      v8f& sa = c ? s10 : s00;
      v8f& sb = c ? s11 : s01;
      v8f& oa = c ? o10 : o00;
      v8f& ob = c ? o11 : o01;
#pragma unroll
      for (int r = 0; r < 8; ++r) {
        float mx = fmaxf(sa[r], sb[r]);
#pragma unroll
        for (int off = 1; off < 16; off <<= 1) mx = fmaxf(mx, __shfl_xor(mx, off, 32));
        const float mnew = fmaxf(mrow[c][r], mx);
        const float corr = __expf(mrow[c][r] - mnew);
        mrow[c][r] = mnew;
        const float p0 = __expf(sa[r] - mnew);
        const float p1 = __expf(sb[r] - mnew);
        float rs = p0 + p1;
#pragma unroll
        for (int off = 1; off < 16; off <<= 1) rs += __shfl_xor(rs, off, 32);
        lrow[c][r] = lrow[c][r] * corr + rs;
        oa[r] *= corr; ob[r] *= corr;
        pbuf[wid][c * 16 + r + lh * 8][lc]      = (_Float16)p0;
        pbuf[wid][c * 16 + r + lh * 8][16 + lc] = (_Float16)p1;
      }
    }
    __syncthreads();  // order LDS transpose writes vs fragment reads
    const v16h pa0 = frag_lds(&pbuf[wid][0][0], 0,  40, lane);
    const v16h pa1 = frag_lds(&pbuf[wid][0][0], 16, 40, lane);
    const v16h vb0 = frag_colmajor(&vsbuf[kt][0], 0,  DHID, lane);
    const v16h vb1 = frag_colmajor(&vsbuf[kt][0], 16, DHID, lane);
    o00 = wmma_f16(pa0, vb0, o00);
    o01 = wmma_f16(pa0, vb1, o01);
    o10 = wmma_f16(pa1, vb0, o10);
    o11 = wmma_f16(pa1, vb1, o11);
    __syncthreads();
  }
  // normalize, gate, store gated output f16 row-major [n*L+l][h*32+d]
#pragma unroll
  for (int c = 0; c < 2; ++c) {
    v8f& oa = c ? o10 : o00;
    v8f& ob = c ? o11 : o01;
#pragma unroll
    for (int r = 0; r < 8; ++r) {
      const int n_idx = row0 + c * 16 + r + lh * 8;
      const float inv = 1.0f / lrow[c][r];
      const size_t mr = (size_t)n_idx * LCOL + li;
#pragma unroll
      for (int s = 0; s < 2; ++s) {
        const int dd = s * 16 + lc;
        const size_t idx = mr * HD + hh * DHID + dd;
        const float val = ((s ? ob[r] : oa[r]) * inv) * (float)gate[idx];
        att[idx] = (_Float16)val;
      }
    }
  }
}

// ------------------------------------------------------- output projection
__global__ __launch_bounds__(256) void msa_outproj_kernel(
    const _Float16* __restrict__ a, const float* __restrict__ wo,
    const float* __restrict__ bo, float* __restrict__ out) {
  __shared__ _Float16 xs[64][40];
  __shared__ _Float16 wt[64][40];
  const int tid   = threadIdx.x;
  const int lane  = tid & 31;
  const int wid   = tid >> 5;
  const int mbase = blockIdx.x * 64;
  const int colt  = blockIdx.y * 64;
  const int lrow  = tid >> 2;
  const int lseg  = (tid & 3) * 8;
  const int frow  = (wid >> 1) * 16;
  const int fcol  = (wid & 1) * 32;
  v8f acc0 = {}, acc1 = {};
  for (int kb = 0; kb < HD; kb += 32) {
    __syncthreads();
    {
      h8 d = *(const h8*)(a + (size_t)(mbase + lrow) * HD + kb + lseg);
#pragma unroll
      for (int j = 0; j < 8; ++j) xs[lrow][lseg + j] = d[j];
    }
    {
      const float* wp = wo + (size_t)(colt + lrow) * HD + kb + lseg;
      float4 d0 = ((const float4*)wp)[0];
      float4 d1 = ((const float4*)wp)[1];
      wt[lrow][lseg + 0] = (_Float16)d0.x; wt[lrow][lseg + 1] = (_Float16)d0.y;
      wt[lrow][lseg + 2] = (_Float16)d0.z; wt[lrow][lseg + 3] = (_Float16)d0.w;
      wt[lrow][lseg + 4] = (_Float16)d1.x; wt[lrow][lseg + 5] = (_Float16)d1.y;
      wt[lrow][lseg + 6] = (_Float16)d1.z; wt[lrow][lseg + 7] = (_Float16)d1.w;
    }
    __syncthreads();
    v16h af = frag_lds(&xs[0][0], frow,      40, lane);
    v16h b0 = frag_lds(&wt[0][0], fcol,      40, lane);
    v16h b1 = frag_lds(&wt[0][0], fcol + 16, 40, lane);
    acc0 = wmma_f16(af, b0, acc0);
    acc1 = wmma_f16(af, b1, acc1);
  }
  const int lh = lane >> 4;
  const int lc = lane & 15;
#pragma unroll
  for (int r = 0; r < 8; ++r) {
    const int m = mbase + frow + r + lh * 8;
#pragma unroll
    for (int s = 0; s < 2; ++s) {
      const int c = colt + fcol + s * 16 + lc;
      const float val = ((s == 0) ? acc0[r] : acc1[r]) + bo[c];
      out[(size_t)m * DM + c] = val;
    }
  }
}

extern "C" void kernel_launch(void* const* d_in, const int* in_sizes, int n_in,
                              void* d_out, int out_size, void* d_ws, size_t ws_size,
                              hipStream_t stream) {
  (void)in_sizes; (void)n_in; (void)out_size; (void)ws_size;
  const float* msa = (const float*)d_in[0];
  const float* lnw = (const float*)d_in[1];
  const float* lnb = (const float*)d_in[2];
  const float* wq  = (const float*)d_in[3];
  const float* wk  = (const float*)d_in[4];
  const float* wv  = (const float*)d_in[5];
  const float* wg  = (const float*)d_in[6];
  const float* bg  = (const float*)d_in[7];
  const float* wo  = (const float*)d_in[8];
  const float* bo  = (const float*)d_in[9];
  float* out = (float*)d_out;

  const size_t sz = (size_t)MROWS * DM;     // halves per f16 buffer (50.3 MB each)
  _Float16* xh = (_Float16*)d_ws;           // LayerNormed x       [m][dm]
  _Float16* qh = xh + sz;                   // Q (pre-scaled)      [l][h][n][d]
  _Float16* kh = qh + sz;                   // K                   [l][h][n][d]
  _Float16* vh = kh + sz;                   // V                   [l][h][n][d]
  _Float16* gh = vh + sz;                   // sigmoid gate        [m][hd]
  _Float16* ah = gh + sz;                   // gated attn output   [m][hd]

  msa_ln_kernel<<<MROWS / 8, 256, 0, stream>>>(msa, lnw, lnb, xh);
  msa_proj_kernel<<<dim3(MROWS / 64, 16), 256, 0, stream>>>(
      xh, wq, wk, wv, wg, bg, qh, kh, vh, gh);
  msa_attn_kernel<<<LCOL * NHEAD, 256, 0, stream>>>(qh, kh, vh, gh, ah);
  msa_outproj_kernel<<<dim3(MROWS / 64, 4), 256, 0, stream>>>(ah, wo, bo, out);
}
